// AttentionBlock_77773267796506
// MI455X (gfx1250) — matrix operand order repaired and checked by reading, not compiled, not run on hardware
//
#include <hip/hip_runtime.h>
#include <hip/hip_bf16.h>
#include <math.h>

typedef __attribute__((ext_vector_type(16))) _Float16 v16h;
typedef __attribute__((ext_vector_type(8)))  _Float16 v8h;
typedef __attribute__((ext_vector_type(8)))  float    v8f;

#define BATCH 8
#define SEQ   1024
#define CH    512
#define NH    8
#define HD    64
#define NQKV  1536   // 3 * NH * HD

// ---------------------------------------------------------------------------
// WMMA fragment helpers (CDNA5 f16 16x16x32, layouts per ISA 7.12.2)
// ---------------------------------------------------------------------------

// A fragment: 16x32 (MxK), f16, source row-major with leading dim ld.
// Lane L: row = L&15, K-halves: [hi*8, hi*8+8) and [16+hi*8, 16+hi*8+8).
template <typename PT>
__device__ inline v16h load_a_frag(PT p, int ld, int m0, int k0, int lane) {
  int r = lane & 15, hi = lane >> 4;
  PT base = p + (size_t)(m0 + r) * ld + k0 + hi * 8;
  v8h lo = *(const v8h*)(base);
  v8h hh = *(const v8h*)(base + 16);
  v16h a;
#pragma unroll
  for (int i = 0; i < 8; ++i) { a[i] = lo[i]; a[i + 8] = hh[i]; }
  return a;
}

// B fragment: 32x16 (KxN), f16. Source stored B-transposed (row n, ld along K)
// K-halves match A: [hi*8, hi*8+8) and [16+hi*8, 16+hi*8+8), two 16B loads.
template <typename PT>
__device__ inline v16h load_b_frag(PT bt, int ld, int k0, int n0, int lane) {
  int n = lane & 15, hi = lane >> 4;
  PT base = bt + (size_t)(n0 + n) * ld + k0 + hi * 8;
  v8h lo = *(const v8h*)(base);
  v8h hh = *(const v8h*)(base + 16);
  v16h b;
#pragma unroll
  for (int i = 0; i < 8; ++i) { b[i] = lo[i]; b[i + 8] = hh[i]; }
  return b;
}

__device__ inline v8f wmma_f16(v16h a, v16h b, v8f c) {
  return __builtin_amdgcn_wmma_f32_16x16x32_f16(false, a, false, b, (short)0, c,
                                                false, false);
}

__device__ inline float grp16_max(float v) {
#pragma unroll
  for (int m = 8; m >= 1; m >>= 1) v = fmaxf(v, __shfl_xor(v, m, 16));
  return v;
}
__device__ inline float grp16_sum(float v) {
#pragma unroll
  for (int m = 8; m >= 1; m >>= 1) v += __shfl_xor(v, m, 16);
  return v;
}

// Async copy: 16 bytes global -> LDS per lane (ASYNCcnt-tracked TDM-lite path)
__device__ inline void async_ld16(uint32_t lds_off, const _Float16* g) {
  asm volatile("global_load_async_to_lds_b128 %0, %1, off"
               :: "v"(lds_off), "v"((uint64_t)(uintptr_t)g)
               : "memory");
}

// ---------------------------------------------------------------------------
// Weight conversion: fp32 -> f16, transposed; fold 1/sqrt(HD) into Q columns.
// ---------------------------------------------------------------------------
__global__ void conv_wqkv_kernel(const float* __restrict__ w, _Float16* __restrict__ wt) {
  int i = blockIdx.x * 256 + threadIdx.x;
  if (i >= NQKV * CH) return;
  int n = i / CH, k = i % CH;           // wt[n][k] = w[k][n]
  float v = w[(size_t)k * NQKV + n];
  if ((n % 192) < HD) v *= 0.125f;      // q columns: fold 1/sqrt(64)
  wt[i] = (_Float16)v;
}

__global__ void conv_wout_kernel(const float* __restrict__ w, _Float16* __restrict__ wt) {
  int i = blockIdx.x * 256 + threadIdx.x;
  if (i >= CH * CH) return;
  int n = i / CH, k = i % CH;
  wt[i] = (_Float16)w[(size_t)k * CH + n];
}

// ---------------------------------------------------------------------------
// GroupNorm: one block per (batch, group). Writes fp32 residual base to out
// and f16 normalized activations for the WMMA GEMMs.
// ---------------------------------------------------------------------------
__global__ __launch_bounds__(256) void groupnorm_kernel(
    const float* __restrict__ x, const float* __restrict__ scale,
    const float* __restrict__ bias, float* __restrict__ xn, _Float16* __restrict__ x16) {
  int b = blockIdx.x >> 5, g = blockIdx.x & 31;
  int tid = threadIdx.x;
  __shared__ float rs[256], rq[256];
  float sum = 0.f, sq = 0.f;
  for (int e = tid; e < 16384; e += 256) {       // 1024 hw * 16 ch
    int hw = e >> 4, cc = e & 15;
    float v = x[((size_t)(b * SEQ + hw)) * CH + g * 16 + cc];
    sum += v; sq += v * v;
  }
  rs[tid] = sum; rq[tid] = sq;
  __syncthreads();
  for (int st = 128; st > 0; st >>= 1) {
    if (tid < st) { rs[tid] += rs[tid + st]; rq[tid] += rq[tid + st]; }
    __syncthreads();
  }
  float mean = rs[0] * (1.f / 16384.f);
  float var  = rq[0] * (1.f / 16384.f) - mean * mean;
  float inv  = rsqrtf(var + 1e-5f);
  for (int e = tid; e < 16384; e += 256) {
    int hw = e >> 4, cc = e & 15, ch = g * 16 + cc;
    size_t idx = ((size_t)(b * SEQ + hw)) * CH + ch;
    float v = (x[idx] - mean) * inv * scale[ch] + bias[ch];
    xn[idx]  = v;
    x16[idx] = (_Float16)v;
  }
}

// ---------------------------------------------------------------------------
// QKV GEMM: [8192,512]f16 @ Wt[1536,512]f16 -> scatter Q[b,h,s,d],
// K[b,h,s,d], V^T[b,h,d,s] as f16. One wave per 16x64 output tile.
// ---------------------------------------------------------------------------
__global__ __launch_bounds__(256) void qkv_gemm_kernel(
    const _Float16* __restrict__ X16, const _Float16* __restrict__ Wt,
    const float* __restrict__ bq, _Float16* __restrict__ Q,
    _Float16* __restrict__ K, _Float16* __restrict__ Vt) {
  int lane = threadIdx.x & 31;
  int wid  = blockIdx.x * 8 + (threadIdx.x >> 5);
  int tn = wid % (NQKV / 64), tm = wid / (NQKV / 64);
  int m0 = tm * 16, n0 = tn * 64;
  int r = lane & 15, hi = lane >> 4;
  v8f acc[4] = {};
  for (int k0 = 0; k0 < CH; k0 += 32) {
    v16h a = load_a_frag(X16, CH, m0, k0, lane);
#pragma unroll
    for (int t = 0; t < 4; ++t)
      acc[t] = wmma_f16(a, load_b_frag(Wt, CH, k0, n0 + 16 * t, lane), acc[t]);
  }
#pragma unroll
  for (int t = 0; t < 4; ++t) {
    int ng = n0 + 16 * t + r;
    int head = ng / 192, within = ng % 192;
    float bias = bq[ng];
    if (within < HD) bias *= 0.125f;   // match folded q scale
#pragma unroll
    for (int j = 0; j < 8; ++j) {
      int row = m0 + 8 * hi + j;
      int b = row >> 10, s = row & 1023;
      int bh = b * NH + head;
      float val = acc[t][j] + bias;
      if (within < HD)
        Q[((size_t)bh * SEQ + s) * HD + within] = (_Float16)val;
      else if (within < 2 * HD)
        K[((size_t)bh * SEQ + s) * HD + (within - HD)] = (_Float16)val;
      else
        Vt[((size_t)bh * HD + (within - 2 * HD)) * SEQ + s] = (_Float16)val;
    }
  }
}

// ---------------------------------------------------------------------------
// Flash attention: 4 waves/block, all sharing one (batch, head). K/V chunks
// are double-buffered into LDS with async global->LDS copies (ASYNCcnt), so
// each 32-key step is loaded once per block instead of once per wave. QK^T
// and PV WMMA B-fragments then come from LDS; S->P relayout stays wave-local.
// ---------------------------------------------------------------------------
#define NKT (SEQ / 32)   // 32 key-chunk iterations

__global__ __launch_bounds__(128) void attn_kernel(
    const _Float16* __restrict__ Q, const _Float16* __restrict__ K,
    const _Float16* __restrict__ Vt, _Float16* __restrict__ O16) {
  __shared__ __attribute__((aligned(32))) _Float16 kbuf[2][32 * 64]; // [key][d]
  __shared__ __attribute__((aligned(32))) _Float16 vbuf[2][64 * 32]; // [d][key]
  __shared__ __attribute__((aligned(32))) _Float16 pbuf[4][16 * 32];
  int tid  = threadIdx.x;
  int lane = tid & 31;
  int w    = tid >> 5;
  int wid  = blockIdx.x * 4 + w;
  int bh   = wid >> 6;              // same for all 4 waves in the block
  int qt   = wid & 63;
  int r = lane & 15, hi = lane >> 4;
  const _Float16* Qb = Q  + (size_t)bh * SEQ * HD;
  const _Float16* Kb = K  + (size_t)bh * SEQ * HD;
  const _Float16* Vb = Vt + (size_t)bh * HD * SEQ;
  _Float16* pb = pbuf[w];

  // stage one 32-key chunk: K rows [key0,key0+32) (4KB) + V^T cols (4KB);
  // 128 threads x 2 chunks x 16B per matrix, all via async LDS copies.
  auto stage = [&](int buf, int key0) {
    uint32_t kls = (uint32_t)(uintptr_t)&kbuf[buf][0];
    uint32_t vls = (uint32_t)(uintptr_t)&vbuf[buf][0];
#pragma unroll
    for (int it = 0; it < 2; ++it) {
      int c = tid + it * 128;
      int krow = c >> 3, kcol = c & 7;      // 32 rows x 8 chunks (128B rows)
      async_ld16(kls + (uint32_t)(krow * 64 + kcol * 8) * 2,
                 Kb + (size_t)(key0 + krow) * HD + kcol * 8);
      int vrow = c >> 2, vcol = c & 3;      // 64 rows x 4 chunks (64B rows)
      async_ld16(vls + (uint32_t)(vrow * 32 + vcol * 8) * 2,
                 Vb + (size_t)vrow * SEQ + key0 + vcol * 8);
    }
  };

  v16h aq0 = load_a_frag(Qb, HD, qt * 16, 0, lane);
  v16h aq1 = load_a_frag(Qb, HD, qt * 16, 32, lane);
  v8f acc[4] = {};
  float mrow[8], lrow[8];
#pragma unroll
  for (int j = 0; j < 8; ++j) { mrow[j] = -INFINITY; lrow[j] = 0.f; }

  stage(0, 0);
  for (int kt = 0; kt < NKT; ++kt) {
    const _Float16* kc = kbuf[kt & 1];
    const _Float16* vc = vbuf[kt & 1];
    if (kt + 1 < NKT) {
      stage((kt + 1) & 1, (kt + 1) * 32);             // prefetch next chunk
      asm volatile("s_wait_asynccnt 4" ::: "memory"); // current chunk landed
    } else {
      asm volatile("s_wait_asynccnt 0" ::: "memory");
    }
    __syncthreads();   // staged chunk visible to all waves

    // S tile 16x32 = two 16x16 WMMA accumulators, K-dim = HD in 2 steps
    v8f s0 = {}, s1 = {};
    s0 = wmma_f16(aq0, load_b_frag(kc, HD, 0,  0,  lane), s0);
    s0 = wmma_f16(aq1, load_b_frag(kc, HD, 32, 0,  lane), s0);
    s1 = wmma_f16(aq0, load_b_frag(kc, HD, 0,  16, lane), s1);
    s1 = wmma_f16(aq1, load_b_frag(kc, HD, 32, 16, lane), s1);
    // online softmax per row (row = 8*hi + j, shared by the 16 lanes of a half)
#pragma unroll
    for (int j = 0; j < 8; ++j) {
      float loc  = grp16_max(fmaxf(s0[j], s1[j]));
      float mnew = fmaxf(mrow[j], loc);
      float sc   = __expf(mrow[j] - mnew);
      float p0   = __expf(s0[j] - mnew);
      float p1   = __expf(s1[j] - mnew);
      lrow[j] = lrow[j] * sc + grp16_sum(p0 + p1);
      mrow[j] = mnew;
#pragma unroll
      for (int t = 0; t < 4; ++t) acc[t][j] *= sc;
      int m = 8 * hi + j;
      pb[m * 32 + r]      = (_Float16)p0;
      pb[m * 32 + r + 16] = (_Float16)p1;
    }
    __builtin_amdgcn_wave_barrier();
    asm volatile("s_wait_dscnt 0" ::: "memory");   // wave-local LDS RAW
    v16h ap = load_a_frag((const _Float16*)pb, 32, 0, 0, lane);
    __builtin_amdgcn_wave_barrier();
    // P(16x32) @ V(32x64): V^T chunk in LDS gives contiguous B fragments
#pragma unroll
    for (int t = 0; t < 4; ++t)
      acc[t] = wmma_f16(ap, load_b_frag(vc, 32, 0, 16 * t, lane), acc[t]);

    __syncthreads();   // all reads of this buffer done before it is re-staged
  }

  int h = bh & (NH - 1), b = bh >> 3;
#pragma unroll
  for (int j = 0; j < 8; ++j) {
    float rl = 1.f / lrow[j];
    int row = qt * 16 + 8 * hi + j;
    size_t base = ((size_t)b * SEQ + row) * CH + h * HD;
#pragma unroll
    for (int t = 0; t < 4; ++t)
      O16[base + 16 * t + r] = (_Float16)(acc[t][j] * rl);
  }
}

// ---------------------------------------------------------------------------
// Output projection + residual: out += O16 @ Wout^T + b_out (fp32 RMW).
// ---------------------------------------------------------------------------
__global__ __launch_bounds__(256) void out_gemm_kernel(
    const _Float16* __restrict__ O16, const _Float16* __restrict__ Wt,
    const float* __restrict__ bo, float* __restrict__ out) {
  int lane = threadIdx.x & 31;
  int wid  = blockIdx.x * 8 + (threadIdx.x >> 5);
  int tn = wid & 7, tm = wid >> 3;   // N tiles = 512/64 = 8
  int m0 = tm * 16, n0 = tn * 64;
  int r = lane & 15, hi = lane >> 4;
  v8f acc[4] = {};
  for (int k0 = 0; k0 < CH; k0 += 32) {
    v16h a = load_a_frag(O16, CH, m0, k0, lane);
#pragma unroll
    for (int t = 0; t < 4; ++t)
      acc[t] = wmma_f16(a, load_b_frag(Wt, CH, k0, n0 + 16 * t, lane), acc[t]);
  }
#pragma unroll
  for (int t = 0; t < 4; ++t) {
    int ng = n0 + 16 * t + r;
    float bias = bo[ng];
#pragma unroll
    for (int j = 0; j < 8; ++j) {
      int row = m0 + 8 * hi + j;
      size_t idx = (size_t)row * CH + ng;
      out[idx] += acc[t][j] + bias;
    }
  }
}

// ---------------------------------------------------------------------------
extern "C" void kernel_launch(void* const* d_in, const int* in_sizes, int n_in,
                              void* d_out, int out_size, void* d_ws, size_t ws_size,
                              hipStream_t stream) {
  const float* x        = (const float*)d_in[0];
  // d_in[1] = t : unused by the reference computation
  const float* gn_scale = (const float*)d_in[2];
  const float* gn_bias  = (const float*)d_in[3];
  const float* w_qkv    = (const float*)d_in[4];
  const float* b_qkv    = (const float*)d_in[5];
  const float* w_out    = (const float*)d_in[6];
  const float* b_out    = (const float*)d_in[7];
  float* out = (float*)d_out;

  char* ws = (char*)d_ws;
  size_t off = 0;
  _Float16* X16  = (_Float16*)(ws + off); off += (size_t)BATCH * SEQ * CH * 2;      // 8 MB
  _Float16* Wq16 = (_Float16*)(ws + off); off += (size_t)NQKV * CH * 2;             // 1.5 MB
  _Float16* Wo16 = (_Float16*)(ws + off); off += (size_t)CH * CH * 2;               // 0.5 MB
  _Float16* Qh   = (_Float16*)(ws + off); off += (size_t)BATCH * NH * SEQ * HD * 2; // 8 MB
  _Float16* Kh   = (_Float16*)(ws + off); off += (size_t)BATCH * NH * SEQ * HD * 2; // 8 MB
  _Float16* Vth  = (_Float16*)(ws + off); off += (size_t)BATCH * NH * HD * SEQ * 2; // 8 MB
  _Float16* O16  = (_Float16*)(ws + off); off += (size_t)BATCH * SEQ * CH * 2;      // 8 MB

  conv_wqkv_kernel<<<(NQKV * CH + 255) / 256, 256, 0, stream>>>(w_qkv, Wq16);
  conv_wout_kernel<<<(CH * CH + 255) / 256, 256, 0, stream>>>(w_out, Wo16);
  groupnorm_kernel<<<BATCH * 32, 256, 0, stream>>>(x, gn_scale, gn_bias, out, X16);
  // 8192/16 M-tiles * 24 N-tiles = 12288 waves / 8 waves-per-block
  qkv_gemm_kernel<<<1536, 256, 0, stream>>>(X16, Wq16, b_qkv, Qh, Kh, Vth);
  // 8*8 heads * 64 q-tiles = 4096 waves / 4 per block
  attn_kernel<<<1024, 128, 0, stream>>>(Qh, Kh, Vth, O16);
  // 512 M-tiles * 8 N-tiles = 4096 waves / 8 per block
  out_gemm_kernel<<<512, 256, 0, stream>>>(O16, Wo16, b_out, out);
}
